// SwinTimeBlock_81475529605589
// MI455X (gfx1250) — matrix operand order repaired and checked by reading, not compiled, not run on hardware
//
#include <hip/hip_runtime.h>
#include <math.h>

// ---------------------------------------------------------------------------
// Problem constants (from reference)
// ---------------------------------------------------------------------------
#define BB   8
#define TT   8
#define HH_  56
#define WW_  56
#define CC   192
#define WS_  7
#define NHH  6
#define LL   (HH_*WW_)           // 3136
#define NWIN 49                  // WS*WS
#define BWIN 512                 // B * (H/WS) * (W/WS)
#define MTOK 200704              // BWIN * T * NWIN  == B*T*L
#define HID_ 768
#define DH   32                  // head dim
#define ROUT 38535168            // B*T*L*C  (one output tensor, fp32 elements)

typedef __bf16 v16bf __attribute__((ext_vector_type(16)));
typedef float  v8f   __attribute__((ext_vector_type(8)));

// ---------------------------------------------------------------------------
// bf16 helpers (raw ushort storage everywhere; bitcast only in device code)
// ---------------------------------------------------------------------------
__device__ __forceinline__ unsigned short f2bf(float f) {
    unsigned u = __float_as_uint(f);
    unsigned r = u + 0x7FFFu + ((u >> 16) & 1u);
    return (unsigned short)(r >> 16);
}
__device__ __forceinline__ float bf2f(unsigned short h) {
    return __uint_as_float(((unsigned)h) << 16);
}
__device__ __forceinline__ v16bf pack_frag(const unsigned short* t) {
    v16bf v; __builtin_memcpy(&v, t, 32); return v;
}

// ---------------------------------------------------------------------------
// CDNA5 async global->LDS copy (ASYNCcnt-tracked), 16 bytes per lane.
// per-lane LDS address in vdst, per-lane 64-bit global address in vaddr (GV mode)
// ---------------------------------------------------------------------------
__device__ __forceinline__ void async_cp16(unsigned lds_addr, const void* gsrc) {
    unsigned long long ga = (unsigned long long)(size_t)gsrc;
    asm volatile("global_load_async_to_lds_b128 %0, %1, off"
                 :: "v"(lds_addr), "v"(ga) : "memory");
}
__device__ __forceinline__ void wait_async0() {
    asm volatile("s_wait_asynccnt 0x0" ::: "memory");
}

// ---------------------------------------------------------------------------
// WMMA fragment loaders — wave32, this file's contiguous k order (same for A and B)
// A (16x32 bf16): lanes 0-15 M=lane, K 0..15;
//                 lanes 16-31 same M, K 16..31.
// ---------------------------------------------------------------------------
__device__ __forceinline__ v16bf load_afrag(const unsigned short* p, int ld) {
    int lane = threadIdx.x & 31, m = lane & 15, g = lane >> 4;
    unsigned short t[16];
#pragma unroll
    for (int i = 0; i < 8; ++i) {
        int kb = 2*i + 16*g;
        t[2*i]   = p[m*ld + kb];
        t[2*i+1] = p[m*ld + kb + 1];
    }
    return pack_frag(t);
}
__device__ __forceinline__ v16bf load_afrag_g(const unsigned short* p, int ld, int rows) {
    int lane = threadIdx.x & 31, m = lane & 15, g = lane >> 4;
    bool ok = (m < rows);
    unsigned short t[16];
#pragma unroll
    for (int i = 0; i < 8; ++i) {
        int kb = 2*i + 16*g;
        t[2*i]   = ok ? p[m*ld + kb]     : (unsigned short)0;
        t[2*i+1] = ok ? p[m*ld + kb + 1] : (unsigned short)0;
    }
    return pack_frag(t);
}
// B (32x16): element(k,n) taken from p[n*ld + k]  (torch-style W[N][K] => B = W^T)
__device__ __forceinline__ v16bf load_bfrag_nk(const unsigned short* p, int ld) {
    int lane = threadIdx.x & 31, n = lane & 15, g = lane >> 4;
    unsigned short t[16];
#pragma unroll
    for (int v = 0; v < 8; ++v) {
        int k = 2*v + 16*g;
        t[2*v]   = p[n*ld + k];
        t[2*v+1] = p[n*ld + k + 1];
    }
    return pack_frag(t);
}
__device__ __forceinline__ v16bf load_bfrag_nk_g(const unsigned short* p, int ld, int rows) {
    int lane = threadIdx.x & 31, n = lane & 15, g = lane >> 4;
    bool ok = (n < rows);
    unsigned short t[16];
#pragma unroll
    for (int v = 0; v < 8; ++v) {
        int k = 2*v + 16*g;
        t[2*v]   = ok ? p[n*ld + k]     : (unsigned short)0;
        t[2*v+1] = ok ? p[n*ld + k + 1] : (unsigned short)0;
    }
    return pack_frag(t);
}
// B (32x16): element(k,n) from p[k*ld + n]  (K-major source, e.g. V rows = keys)
__device__ __forceinline__ v16bf load_bfrag_kn_g(const unsigned short* p, int ld, int krows) {
    int lane = threadIdx.x & 31, n = lane & 15, g = lane >> 4;
    unsigned short t[16];
#pragma unroll
    for (int v = 0; v < 8; ++v) {
        int k = 2*v + 16*g;
        t[2*v]   = (k     < krows) ? p[(k    )*ld + n] : (unsigned short)0;
        t[2*v+1] = (k + 1 < krows) ? p[(k + 1)*ld + n] : (unsigned short)0;
    }
    return pack_frag(t);
}
__device__ __forceinline__ v8f wmma_bf16(v16bf a, v16bf b, v8f c) {
    return __builtin_amdgcn_wmma_f32_16x16x32_bf16(false, a, false, b, (short)0, c, false, false);
}

// windowed token row -> flat (b,t,l) token index (window_reverse)
__device__ __forceinline__ size_t map_row(int row) {
    int b_  = row / (TT * NWIN);
    int rem = row - b_ * (TT * NWIN);
    int t   = rem / NWIN;
    int n   = rem - t * NWIN;
    int bb  = b_ >> 6;            // b
    int hw  = b_ & 63;
    int hh  = hw >> 3, ww = hw & 7;
    int r   = n / WS_, s = n - r * WS_;
    int l   = (hh*WS_ + r) * WW_ + (ww*WS_ + s);
    return ((size_t)(bb*TT + t)) * LL + l;
}

// ---------------------------------------------------------------------------
// fp32 -> bf16 conversion (weights + activations)
// ---------------------------------------------------------------------------
__global__ void k_cvt(const float* __restrict__ src, unsigned short* __restrict__ dst, int n) {
    int i = blockIdx.x * 256 + threadIdx.x;
    if (i < n) dst[i] = f2bf(src[i]);
}

// ---------------------------------------------------------------------------
// LN1 + window partition: x (B,T,L,C) fp32 -> xn (MTOK, C) bf16 windowed order
// ---------------------------------------------------------------------------
__global__ void __launch_bounds__(256) k_ln1(const float* __restrict__ x,
                                             const float* __restrict__ gg,
                                             const float* __restrict__ bb,
                                             unsigned short* __restrict__ xn) {
    int lane = threadIdx.x & 31;
    int row  = blockIdx.x * 8 + (threadIdx.x >> 5);
    const float* xr = x + map_row(row) * CC;
    float v[6], sm = 0.f;
#pragma unroll
    for (int e = 0; e < 6; ++e) { v[e] = xr[e*32 + lane]; sm += v[e]; }
#pragma unroll
    for (int off = 16; off >= 1; off >>= 1) sm += __shfl_xor(sm, off);
    float mean = sm * (1.0f / CC);
    float vs = 0.f;
#pragma unroll
    for (int e = 0; e < 6; ++e) { float d = v[e] - mean; vs += d*d; }
#pragma unroll
    for (int off = 16; off >= 1; off >>= 1) vs += __shfl_xor(vs, off);
    float inv = rsqrtf(vs * (1.0f / CC) + 1e-5f);
    unsigned short* o = xn + (size_t)row * CC;
#pragma unroll
    for (int e = 0; e < 6; ++e) {
        int c = e*32 + lane;
        o[c] = f2bf((v[e] - mean) * inv * gg[c] + bb[c]);
    }
}

// ---------------------------------------------------------------------------
// GEMM core: block 256 threads = 8 waves (4x2), block tile 128x64, K step 32.
// Double-buffered LDS tiles filled with async global->LDS copies (ASYNCcnt).
// out = A(M x K, bf16) @ W(N x K, bf16)^T   — accumulates into acc[2][2]
// ---------------------------------------------------------------------------
#define ASZ (128*40)
#define BSZ (64*40)

__device__ __forceinline__ void stage_tile(const unsigned short* __restrict__ A, int lda,
                                           const unsigned short* __restrict__ W, int ldw,
                                           size_t m0, int n0, int k0,
                                           unsigned short* As, unsigned short* Bs) {
    int tid = threadIdx.x;
    {   // A tile 128x32 : two b128 per thread (rows of 80B in LDS, 16B aligned)
        int row = tid >> 1, kk = (tid & 1) * 16;
        const unsigned short* s = A + (m0 + row) * (size_t)lda + k0 + kk;
        unsigned d = (unsigned)(size_t)(As + row*40 + kk);
        async_cp16(d,      s);
        async_cp16(d + 16, s + 8);
    }
    {   // B tile 64 rows x 32 k : one b128 per thread
        int n = tid >> 2, kk = (tid & 3) * 8;
        const unsigned short* s = W + (size_t)(n0 + n) * ldw + k0 + kk;
        unsigned d = (unsigned)(size_t)(Bs + n*40 + kk);
        async_cp16(d, s);
    }
}

__device__ __forceinline__ void gemm_core(const unsigned short* __restrict__ A, int lda,
                                          const unsigned short* __restrict__ W, int ldw,
                                          int K, size_t m0, int n0,
                                          unsigned short* As, unsigned short* Bs, // 2 buffers each
                                          v8f acc[2][2]) {
    int wave = threadIdx.x >> 5, wm = wave & 3, wn = wave >> 2;
    int nk = K >> 5;
    stage_tile(A, lda, W, ldw, m0, n0, 0, As, Bs);
    for (int kt = 0; kt < nk; ++kt) {
        unsigned short* Ab = As + (kt & 1) * ASZ;
        unsigned short* Bb = Bs + (kt & 1) * BSZ;
        wait_async0();          // this wave's copies for buffer kt&1 landed
        __syncthreads();        // everyone's copies landed; prev buffer fully read
        if (kt + 1 < nk)
            stage_tile(A, lda, W, ldw, m0, n0, (kt + 1) * 32,
                       As + ((kt + 1) & 1) * ASZ, Bs + ((kt + 1) & 1) * BSZ);
        v16bf a0 = load_afrag(Ab + (wm*32 +  0)*40, 40);
        v16bf a1 = load_afrag(Ab + (wm*32 + 16)*40, 40);
        v16bf b0 = load_bfrag_nk(Bb + (wn*32 +  0)*40, 40);
        v16bf b1 = load_bfrag_nk(Bb + (wn*32 + 16)*40, 40);
        acc[0][0] = wmma_bf16(a0, b0, acc[0][0]);
        acc[0][1] = wmma_bf16(a0, b1, acc[0][1]);
        acc[1][0] = wmma_bf16(a1, b0, acc[1][0]);
        acc[1][1] = wmma_bf16(a1, b1, acc[1][1]);
    }
    __syncthreads();            // protect buffers before any caller re-stages
}

// QKV: out bf16 (M x Nout), + bias
__global__ void __launch_bounds__(256) k_gemm_bf16(const unsigned short* __restrict__ A,
                                                   const unsigned short* __restrict__ W,
                                                   const float* __restrict__ bias,
                                                   unsigned short* __restrict__ out,
                                                   int K, int Nout) {
    __shared__ unsigned short As[2*ASZ];
    __shared__ unsigned short Bs[2*BSZ];
    size_t m0 = (size_t)blockIdx.x * 128;
    int    n0 = blockIdx.y * 64;
    v8f acc[2][2] = {};
    gemm_core(A, K, W, K, K, m0, n0, As, Bs, acc);
    int lane = threadIdx.x & 31, n15 = lane & 15, g = lane >> 4;
    int wave = threadIdx.x >> 5, wm = wave & 3, wn = wave >> 2;
#pragma unroll
    for (int ti = 0; ti < 2; ++ti)
#pragma unroll
        for (int tj = 0; tj < 2; ++tj) {
            int col = n0 + wn*32 + tj*16 + n15;
            float bia = bias[col];
            size_t rowb = m0 + wm*32 + ti*16 + 8*g;
#pragma unroll
            for (int r = 0; r < 8; ++r)
                out[(rowb + r) * (size_t)Nout + col] = f2bf(acc[ti][tj][r] + bia);
        }
}

// projection + window_reverse + shortcut, fp32 out (N fixed = 192)
__global__ void __launch_bounds__(256) k_gemm_proj(const unsigned short* __restrict__ A,
                                                   const unsigned short* __restrict__ W,
                                                   const float* __restrict__ bias,
                                                   const float* __restrict__ shortcut,
                                                   float* __restrict__ outF) {
    __shared__ unsigned short As[2*ASZ];
    __shared__ unsigned short Bs[2*BSZ];
    size_t m0 = (size_t)blockIdx.x * 128;
    int    n0 = blockIdx.y * 64;
    v8f acc[2][2] = {};
    gemm_core(A, CC, W, CC, CC, m0, n0, As, Bs, acc);
    int lane = threadIdx.x & 31, n15 = lane & 15, g = lane >> 4;
    int wave = threadIdx.x >> 5, wm = wave & 3, wn = wave >> 2;
#pragma unroll
    for (int ti = 0; ti < 2; ++ti)
#pragma unroll
        for (int tj = 0; tj < 2; ++tj) {
            int col = n0 + wn*32 + tj*16 + n15;
            float bia = bias[col];
#pragma unroll
            for (int r = 0; r < 8; ++r) {
                int row = (int)m0 + wm*32 + ti*16 + r + 8*g;
                size_t idx = map_row(row) * CC + col;
                outF[idx] = shortcut[idx] + acc[ti][tj][r] + bia;
            }
        }
}

// fuse: out = [x_t | x_sp] @ fuse_w^T + b  (two K=192 passes, fp32 out)
__global__ void __launch_bounds__(256) k_gemm_fuse(const unsigned short* __restrict__ At,
                                                   const unsigned short* __restrict__ Asp,
                                                   const unsigned short* __restrict__ Wf,
                                                   const float* __restrict__ bias,
                                                   float* __restrict__ outF) {
    __shared__ unsigned short As[2*ASZ];
    __shared__ unsigned short Bs[2*BSZ];
    size_t m0 = (size_t)blockIdx.x * 128;
    int    n0 = blockIdx.y * 64;
    v8f acc[2][2] = {};
    gemm_core(At,  CC, Wf,      2*CC, CC, m0, n0, As, Bs, acc);
    gemm_core(Asp, CC, Wf + CC, 2*CC, CC, m0, n0, As, Bs, acc);
    int lane = threadIdx.x & 31, n15 = lane & 15, g = lane >> 4;
    int wave = threadIdx.x >> 5, wm = wave & 3, wn = wave >> 2;
#pragma unroll
    for (int ti = 0; ti < 2; ++ti)
#pragma unroll
        for (int tj = 0; tj < 2; ++tj) {
            int col = n0 + wn*32 + tj*16 + n15;
            float bia = bias[col];
            size_t rowb = m0 + wm*32 + ti*16 + 8*g;
#pragma unroll
            for (int r = 0; r < 8; ++r)
                outF[(rowb + r) * (size_t)CC + col] = acc[ti][tj][r] + bia;
        }
}

// ---------------------------------------------------------------------------
// Spatial windowed attention: one wave per (window-frame, head).
// 49 tokens padded to 64; d=32 => one WMMA K-step for QK^T, two for P.V.
// ---------------------------------------------------------------------------
__global__ void __launch_bounds__(128) k_attn_sp(const unsigned short* __restrict__ qkv,
                                                 unsigned short* __restrict__ out) {
    __shared__ unsigned short Pl[4][64*72];
    int wave = threadIdx.x >> 5;
    int gid  = blockIdx.x * 4 + wave;        // 0..24575
    int h    = gid % NHH;
    int inst = gid / NHH;                    // (b_,t) flattened, 0..4095
    const unsigned short* base = qkv + (size_t)inst * NWIN * (3*CC);
    int qoff = h*DH, koff = CC + h*DH, voff = 2*CC + h*DH;
    int lane = threadIdx.x & 31, n15 = lane & 15, g = lane >> 4;

    v16bf qf[4];
#pragma unroll
    for (int i = 0; i < 4; ++i)
        qf[i] = load_afrag_g(base + (size_t)(i*16)*(3*CC) + qoff, 3*CC, NWIN - i*16);

    v8f S[4][4];
#pragma unroll
    for (int j = 0; j < 4; ++j) {
        v16bf kf = load_bfrag_nk_g(base + (size_t)(j*16)*(3*CC) + koff, 3*CC, NWIN - j*16);
#pragma unroll
        for (int i = 0; i < 4; ++i) {
            v8f z = {};
            S[i][j] = wmma_bf16(qf[i], kf, z);
        }
    }
    const float sc = 0.17677669529663687f;   // 1/sqrt(32)
#pragma unroll
    for (int i = 0; i < 4; ++i)
#pragma unroll
        for (int j = 0; j < 4; ++j)
#pragma unroll
            for (int r = 0; r < 8; ++r) {
                float v = S[i][j][r] * sc;
                if (j == 3 && n15 > 0) v = -1e30f;   // cols 49..63 masked
                S[i][j][r] = v;
            }
    // row softmax: row lives in one 16-lane half, spread over 4 col tiles
#pragma unroll
    for (int i = 0; i < 4; ++i)
#pragma unroll
        for (int r = 0; r < 8; ++r) {
            float mx = fmaxf(fmaxf(S[i][0][r], S[i][1][r]), fmaxf(S[i][2][r], S[i][3][r]));
#pragma unroll
            for (int off = 8; off >= 1; off >>= 1) mx = fmaxf(mx, __shfl_xor(mx, off));
            float sum = 0.f;
#pragma unroll
            for (int j = 0; j < 4; ++j) {
                float e = __expf(S[i][j][r] - mx);
                S[i][j][r] = e; sum += e;
            }
#pragma unroll
            for (int off = 8; off >= 1; off >>= 1) sum += __shfl_xor(sum, off);
            float inv = 1.0f / sum;
#pragma unroll
            for (int j = 0; j < 4; ++j) S[i][j][r] *= inv;
        }
    // stage P (64x64) in LDS to re-fragment as A operand
    unsigned short* P = Pl[wave];
#pragma unroll
    for (int i = 0; i < 4; ++i)
#pragma unroll
        for (int j = 0; j < 4; ++j)
#pragma unroll
            for (int r = 0; r < 8; ++r)
                P[(i*16 + r + 8*g)*72 + j*16 + n15] = f2bf(S[i][j][r]);

    v8f O[4][2] = {};
#pragma unroll
    for (int a = 0; a < 2; ++a) {
        v16bf vf[2];
#pragma unroll
        for (int jn = 0; jn < 2; ++jn)
            vf[jn] = load_bfrag_kn_g(base + (size_t)(a*32)*(3*CC) + voff + jn*16,
                                     3*CC, NWIN - a*32);
#pragma unroll
        for (int i = 0; i < 4; ++i) {
            v16bf pa = load_afrag(P + (i*16)*72 + a*32, 72);
#pragma unroll
            for (int jn = 0; jn < 2; ++jn) O[i][jn] = wmma_bf16(pa, vf[jn], O[i][jn]);
        }
    }
#pragma unroll
    for (int i = 0; i < 4; ++i)
#pragma unroll
        for (int r = 0; r < 8; ++r) {
            int m = i*16 + r + 8*g;
            if (m < NWIN) {
#pragma unroll
                for (int jn = 0; jn < 2; ++jn)
                    out[((size_t)inst * NWIN + m) * CC + h*DH + jn*16 + n15] = f2bf(O[i][jn][r]);
            }
        }
}

// ---------------------------------------------------------------------------
// Temporal attention (T=8): one thread per (window-pixel, head, query)
// ---------------------------------------------------------------------------
__global__ void __launch_bounds__(256) k_attn_t(const unsigned short* __restrict__ qkv,
                                                unsigned short* __restrict__ out) {
    int tid = blockIdx.x * 256 + threadIdx.x;        // 1,204,224 threads
    int q   = tid & 7;
    int h   = (tid >> 3) % NHH;
    int pix = tid / (8*NHH);                         // 0..25087
    int b_  = pix / NWIN, n = pix % NWIN;
    size_t ib = ((size_t)b_ * TT) * NWIN + n;        // row for t=0; row(t)=ib+t*49
    const unsigned short* Q = qkv + (ib + (size_t)q*NWIN) * (3*CC) + h*DH;
    float qv[DH];
#pragma unroll
    for (int k = 0; k < DH; ++k) qv[k] = bf2f(Q[k]);
    float scl = 0.17677669529663687f;
    float sc[TT], mx = -1e30f;
    for (int t = 0; t < TT; ++t) {
        const unsigned short* K = qkv + (ib + (size_t)t*NWIN) * (3*CC) + CC + h*DH;
        float d = 0.f;
#pragma unroll
        for (int k = 0; k < DH; ++k) d += qv[k] * bf2f(K[k]);
        d *= scl; sc[t] = d; mx = fmaxf(mx, d);
    }
    float sum = 0.f;
    for (int t = 0; t < TT; ++t) { sc[t] = __expf(sc[t] - mx); sum += sc[t]; }
    float inv = 1.0f / sum;
    float o[DH] = {};
    for (int t = 0; t < TT; ++t) {
        float p = sc[t] * inv;
        const unsigned short* V = qkv + (ib + (size_t)t*NWIN) * (3*CC) + 2*CC + h*DH;
#pragma unroll
        for (int k = 0; k < DH; ++k) o[k] += p * bf2f(V[k]);
    }
    unsigned short* O = out + (ib + (size_t)q*NWIN) * CC + h*DH;
#pragma unroll
    for (int k = 0; k < DH; ++k) O[k] = f2bf(o[k]);
}

// ---------------------------------------------------------------------------
// Fused LN2 + MLP (fc1 -> exact GELU -> fc2) + residual, in-place on X.
// 32 token rows per block; hidden (32x768 bf16) lives in LDS.
// ---------------------------------------------------------------------------
__global__ void __launch_bounds__(256) k_ln2_mlp(float* __restrict__ X,
                                                 const float* __restrict__ gg,
                                                 const float* __restrict__ bb,
                                                 const unsigned short* __restrict__ W1,
                                                 const float* __restrict__ b1,
                                                 const unsigned short* __restrict__ W2,
                                                 const float* __restrict__ b2) {
    __shared__ unsigned short xn_s[32*200];
    __shared__ unsigned short h_s[32*776];
    size_t row0 = (size_t)blockIdx.x * 32;
    int tid = threadIdx.x;
    if (tid < 32) {                       // LayerNorm, one thread per row
        const float* xr = X + (row0 + tid) * CC;
        float m = 0.f;
        for (int c = 0; c < CC; ++c) m += xr[c];
        m *= (1.0f / CC);
        float v = 0.f;
        for (int c = 0; c < CC; ++c) { float d = xr[c] - m; v += d*d; }
        float inv = rsqrtf(v * (1.0f / CC) + 1e-5f);
        for (int c = 0; c < CC; ++c)
            xn_s[tid*200 + c] = f2bf((xr[c] - m) * inv * gg[c] + bb[c]);
    }
    __syncthreads();
    int lane = tid & 31, n15 = lane & 15, g = lane >> 4;
    int wave = tid >> 5;
    // GEMM1: (32 x 768) = xn @ W1^T ; wave w owns 96 output cols
    for (int ct = 0; ct < 6; ++ct) {
        int c0 = wave*96 + ct*16;
#pragma unroll
        for (int rt = 0; rt < 2; ++rt) {
            v8f acc = {};
            for (int k0 = 0; k0 < CC; k0 += 32) {
                v16bf a   = load_afrag(xn_s + rt*16*200 + k0, 200);
                v16bf bfr = load_bfrag_nk(W1 + (size_t)c0*CC + k0, CC);
                acc = wmma_bf16(a, bfr, acc);
            }
#pragma unroll
            for (int r = 0; r < 8; ++r) {
                int m = rt*16 + r + 8*g, n = c0 + n15;
                float xv = acc[r] + b1[n];
                float ge = 0.5f * xv * (1.0f + erff(xv * 0.70710678118654752f));
                h_s[m*776 + n] = f2bf(ge);
            }
        }
    }
    __syncthreads();
    // GEMM2: (32 x 192) = h @ W2^T ; 24 tiles, 3 per wave
#pragma unroll
    for (int s = 0; s < 3; ++s) {
        int t  = wave*3 + s;
        int rt = t & 1, ct = t >> 1;      // ct 0..11
        int c0 = ct*16;
        v8f acc = {};
        for (int k0 = 0; k0 < HID_; k0 += 32) {
            v16bf a   = load_afrag(h_s + rt*16*776 + k0, 776);
            v16bf bfr = load_bfrag_nk(W2 + (size_t)c0*HID_ + k0, HID_);
            acc = wmma_bf16(a, bfr, acc);
        }
#pragma unroll
        for (int r = 0; r < 8; ++r) {
            int m = rt*16 + r + 8*g, n = c0 + n15;
            X[(row0 + m)*CC + n] += acc[r] + b2[n];
        }
    }
}

// ---------------------------------------------------------------------------
// Host launch
// ---------------------------------------------------------------------------
extern "C" void kernel_launch(void* const* d_in, const int* in_sizes, int n_in,
                              void* d_out, int out_size, void* d_ws, size_t ws_size,
                              hipStream_t stream) {
    (void)in_sizes; (void)n_in; (void)out_size; (void)ws_size;
    const float* x     = (const float*)d_in[0];
    const float* n1g   = (const float*)d_in[2];
    const float* n1b   = (const float*)d_in[3];
    const float* qkvw  = (const float*)d_in[4];
    const float* qkvb  = (const float*)d_in[5];
    const float* pspw  = (const float*)d_in[6];
    const float* pspb  = (const float*)d_in[7];
    const float* ptw   = (const float*)d_in[8];
    const float* ptb   = (const float*)d_in[9];
    const float* n2g   = (const float*)d_in[10];
    const float* n2b   = (const float*)d_in[11];
    const float* t1w   = (const float*)d_in[12];
    const float* t1b   = (const float*)d_in[13];
    const float* t2w   = (const float*)d_in[14];
    const float* t2b   = (const float*)d_in[15];
    const float* s1w   = (const float*)d_in[16];
    const float* s1b   = (const float*)d_in[17];
    const float* s2w   = (const float*)d_in[18];
    const float* s2b   = (const float*)d_in[19];
    const float* fw    = (const float*)d_in[20];
    const float* fb    = (const float*)d_in[21];

    float* outF = (float*)d_out;              // out  (M x 192)
    float* xt   = outF + (size_t)ROUT;        // x_t  (M x 192, flat order)
    float* xsp  = outF + (size_t)2*ROUT;      // x_sp

    char* ws = (char*)d_ws;
    size_t off = 0;
    auto alloc = [&](size_t bytes) -> void* {
        void* p = ws + off;
        off += (bytes + 255) & ~(size_t)255;
        return p;
    };
    unsigned short* wq  = (unsigned short*)alloc((size_t)3*CC*CC * 2);     // 576x192
    unsigned short* wps = (unsigned short*)alloc((size_t)CC*CC * 2);
    unsigned short* wpt = (unsigned short*)alloc((size_t)CC*CC * 2);
    unsigned short* wt1 = (unsigned short*)alloc((size_t)HID_*CC * 2);
    unsigned short* wt2 = (unsigned short*)alloc((size_t)CC*HID_ * 2);
    unsigned short* ws1 = (unsigned short*)alloc((size_t)HID_*CC * 2);
    unsigned short* ws2 = (unsigned short*)alloc((size_t)CC*HID_ * 2);
    unsigned short* wf  = (unsigned short*)alloc((size_t)CC*2*CC * 2);
    unsigned short* bufA = (unsigned short*)alloc((size_t)MTOK*CC * 2);    // xn / attn_sp / xsp_bf
    unsigned short* qkv  = (unsigned short*)alloc((size_t)MTOK*3*CC * 2);  // QKV bf16
    unsigned short* bufB = (unsigned short*)alloc((size_t)MTOK*CC * 2);    // attn_t / xt_bf

    auto cvt = [&](const float* s, unsigned short* d, int n) {
        k_cvt<<<(n + 255) / 256, 256, 0, stream>>>(s, d, n);
    };
    cvt(qkvw, wq,  3*CC*CC);
    cvt(pspw, wps, CC*CC);
    cvt(ptw,  wpt, CC*CC);
    cvt(t1w,  wt1, HID_*CC);
    cvt(t2w,  wt2, CC*HID_);
    cvt(s1w,  ws1, HID_*CC);
    cvt(s2w,  ws2, CC*HID_);
    cvt(fw,   wf,  CC*2*CC);

    // 1) LN1 + window partition -> bufA (bf16, windowed token order)
    k_ln1<<<MTOK/8, 256, 0, stream>>>(x, n1g, n1b, bufA);

    // 2) QKV GEMM (shared by spatial & temporal attention)
    k_gemm_bf16<<<dim3(MTOK/128, (3*CC)/64), 256, 0, stream>>>(bufA, wq, qkvb, qkv, CC, 3*CC);

    // 3) spatial attention -> bufA (xn dead), temporal -> bufB
    k_attn_sp<<<(4096*NHH)/4, 128, 0, stream>>>(qkv, bufA);
    k_attn_t<<<(25088*NHH*TT)/256, 256, 0, stream>>>(qkv, bufB);

    // 4) projections + window_reverse + shortcut -> x_t / x_sp (fp32 in d_out)
    k_gemm_proj<<<dim3(MTOK/128, CC/64), 256, 0, stream>>>(bufB, wpt, ptb, x, xt);
    k_gemm_proj<<<dim3(MTOK/128, CC/64), 256, 0, stream>>>(bufA, wps, pspb, x, xsp);

    // 5) fused LN2 + MLP + residual (in place)
    k_ln2_mlp<<<MTOK/32, 256, 0, stream>>>(xt,  n2g, n2b, wt1, t1b, wt2, t2b);
    k_ln2_mlp<<<MTOK/32, 256, 0, stream>>>(xsp, n2g, n2b, ws1, s1b, ws2, s2b);

    // 6) fuse: out = [x_t | x_sp] @ fuse_w^T + b
    cvt(xt,  bufB, MTOK*CC);
    cvt(xsp, bufA, MTOK*CC);
    k_gemm_fuse<<<dim3(MTOK/128, CC/64), 256, 0, stream>>>(bufB, bufA, wf, fb, outF);
}
